// SAMVisionAttention_9904194584921
// MI455X (gfx1250) — matrix operand order repaired and checked by reading, not compiled, not run on hardware
//
#include <hip/hip_runtime.h>
#include <hip/hip_bf16.h>

typedef __bf16 bf16_t;
typedef __attribute__((ext_vector_type(16))) __bf16 v16bf;
typedef __attribute__((ext_vector_type(8)))  float   v8f;

// ---------------------------------------------------------------------------
// WMMA helpers (CDNA5: D = A(16x32 bf16) * B(32x16 bf16) + C(16x16 f32))
// ---------------------------------------------------------------------------
__device__ __forceinline__ v8f wmma_bf16(v16bf a, v16bf b, v8f c) {
  return __builtin_amdgcn_wmma_f32_16x16x32_bf16(
      /*neg_a=*/false, a, /*neg_b=*/false, b,
      /*c_mod=*/(short)0, c, /*reuse_a=*/false, /*reuse_b=*/false);
}

union Frag { v16bf v; uint4 q[2]; };

// A fragment, 16x32 bf16, row-major source, leading dim `lda` (elements).
__device__ __forceinline__ v16bf load_a_frag(const bf16_t* A, int lda,
                                             int row0, int k0, int lane) {
  const int r  = row0 + (lane & 15);
  const int ks = (lane >> 4) << 3;
  const bf16_t* p = A + (size_t)r * lda + k0 + ks;
  Frag f;
  f.q[0] = *(const uint4*)(p);
  f.q[1] = *(const uint4*)(p + 16);
  return f.v;
}

// B fragment, 32x16 bf16 (KxN) from B^T row-major: Bt[n][k], leading dim ldb.
__device__ __forceinline__ v16bf load_b_frag(const bf16_t* Bt, int ldb,
                                             int n0, int k0, int lane) {
  const int n = n0 + (lane & 15);
  const int k = k0 + ((lane >> 4) << 3);
  const bf16_t* p = Bt + (size_t)n * ldb + k;
  Frag f;
  f.q[0] = *(const uint4*)(p);
  f.q[1] = *(const uint4*)(p + 16);
  return f.v;
}

// ---------------------------------------------------------------------------
// CDNA5 async global->LDS copy (ASYNCcnt-tracked), per-lane 16B.
// ---------------------------------------------------------------------------
__device__ __forceinline__ void async_b128(void* lds_dst, const void* gsrc) {
  const unsigned loff = (unsigned)(uintptr_t)lds_dst;  // low 32b = LDS offset
  asm volatile("global_load_async_to_lds_b128 %0, %1, off"
               :: "v"(loff), "v"(gsrc) : "memory");
}
__device__ __forceinline__ void wait_async0() {
  asm volatile("s_wait_asynccnt 0" ::: "memory");
}

// Copy one 64x64 bf16 tile (64 rows x 128B) into LDS (pitch 64 elements).
__device__ __forceinline__ void copy_tile_async(bf16_t* dstLds,
                                                const bf16_t* srcG,
                                                int rowStrideG, int tid) {
  for (int c = tid; c < 512; c += 128) {
    const int row = c >> 3, seg = c & 7;
    async_b128(dstLds + row * 64 + seg * 8,
               srcG + (size_t)row * rowStrideG + seg * 8);
  }
}

// ---------------------------------------------------------------------------
// Prep kernels
// ---------------------------------------------------------------------------
__global__ void cvt_bf16(const float* __restrict__ in, bf16_t* __restrict__ out, int n) {
  int i = blockIdx.x * blockDim.x + threadIdx.x;
  if (i < n) out[i] = (bf16_t)in[i];
}

__global__ void transpose_cvt(const float* __restrict__ in, bf16_t* __restrict__ out,
                              int rows, int cols) {
  int i = blockIdx.x * blockDim.x + threadIdx.x;
  if (i < rows * cols) {
    int c = i / rows, r = i % rows;
    out[i] = (bf16_t)in[r * cols + c];
  }
}

__global__ void build_rel(const float* __restrict__ rp, bf16_t* __restrict__ out, int n) {
  int i = blockIdx.x * blockDim.x + threadIdx.x;
  if (i < n) {
    int c = i & 63, j = (i >> 6) & 63, q = i >> 12;
    out[i] = (bf16_t)rp[(q - j + 63) * 64 + c];
  }
}

// ---------------------------------------------------------------------------
// GEMM core: 64x64 tile per block (4 waves x 16 cols), K pipelined with two
// named fragment sets (no register ping-pong copies). KTOT must be /64.
// ---------------------------------------------------------------------------
template <int KTOT>
__device__ __forceinline__ void gemm_tile(const bf16_t* __restrict__ Ag, int lda,
                                          const bf16_t* __restrict__ Bt, int ldb,
                                          int m0, int n0, int lane, v8f acc[4]) {
  v16bf aA[4], aB[4], bA, bB;
  bA = load_b_frag(Bt, ldb, n0, 0, lane);
#pragma unroll
  for (int mt = 0; mt < 4; ++mt) aA[mt] = load_a_frag(Ag, lda, m0 + mt * 16, 0, lane);
#pragma unroll 1
  for (int s = 0; s < KTOT - 64; s += 64) {
    bB = load_b_frag(Bt, ldb, n0, s + 32, lane);
#pragma unroll
    for (int mt = 0; mt < 4; ++mt) aB[mt] = load_a_frag(Ag, lda, m0 + mt * 16, s + 32, lane);
#pragma unroll
    for (int mt = 0; mt < 4; ++mt) acc[mt] = wmma_bf16(aA[mt], bA, acc[mt]);
    bA = load_b_frag(Bt, ldb, n0, s + 64, lane);
#pragma unroll
    for (int mt = 0; mt < 4; ++mt) aA[mt] = load_a_frag(Ag, lda, m0 + mt * 16, s + 64, lane);
#pragma unroll
    for (int mt = 0; mt < 4; ++mt) acc[mt] = wmma_bf16(aB[mt], bB, acc[mt]);
  }
  bB = load_b_frag(Bt, ldb, n0, KTOT - 32, lane);
#pragma unroll
  for (int mt = 0; mt < 4; ++mt) aB[mt] = load_a_frag(Ag, lda, m0 + mt * 16, KTOT - 32, lane);
#pragma unroll
  for (int mt = 0; mt < 4; ++mt) acc[mt] = wmma_bf16(aA[mt], bA, acc[mt]);
#pragma unroll
  for (int mt = 0; mt < 4; ++mt) acc[mt] = wmma_bf16(aB[mt], bB, acc[mt]);
}

// ---------------------------------------------------------------------------
// QKV GEMM: [4096,768] x [768,2304] + bias -> per-head q,k (row-major), v^T.
// ---------------------------------------------------------------------------
__global__ __launch_bounds__(128) void qkv_gemm(
    const bf16_t* __restrict__ xb, const bf16_t* __restrict__ wt,
    const float* __restrict__ bias,
    bf16_t* __restrict__ qb, bf16_t* __restrict__ kb, bf16_t* __restrict__ vtb) {
  const int tid = threadIdx.x, lane = tid & 31, wv = tid >> 5;
  const int m0 = blockIdx.x * 64;
  const int n0 = blockIdx.y * 64 + (wv << 4);
  v8f acc[4] = {};
  gemm_tile<768>(xb, 768, wt, 768, m0, n0, lane, acc);

  const int n = n0 + (lane & 15);
  const float bn = bias[n];
  const int t = n / 768, rem = n % 768, head = rem >> 6, d = rem & 63;
  const int hi8 = (lane >> 4) << 3;
#pragma unroll
  for (int mt = 0; mt < 4; ++mt) {
#pragma unroll
    for (int r = 0; r < 8; ++r) {
      const int m = m0 + mt * 16 + r + hi8;
      const bf16_t bv = (bf16_t)(acc[mt][r] + bn);
      if (t == 0)      qb[((size_t)head * 4096 + m) * 64 + d] = bv;
      else if (t == 1) kb[((size_t)head * 4096 + m) * 64 + d] = bv;
      else             vtb[((size_t)head * 64 + d) * 4096 + m] = bv;
    }
  }
}

// ---------------------------------------------------------------------------
// Rel-pos bias GEMMs (64x64x64 each); all fragments preloaded.
// ---------------------------------------------------------------------------
__global__ __launch_bounds__(128) void relbias_gemm(
    const bf16_t* __restrict__ qb, const bf16_t* __restrict__ Rh,
    const bf16_t* __restrict__ Rw, float* __restrict__ relh,
    float* __restrict__ relw) {
  const int tid = threadIdx.x, lane = tid & 31, wv = tid >> 5;
  const int i = blockIdx.x;   // h (z=0) or w (z=1)
  const int b = blockIdx.y;   // head
  const bf16_t* A; const bf16_t* B; float* out; int lda, orow;
  if (blockIdx.z == 0) {
    A = qb + ((size_t)b * 4096 + i * 64) * 64; lda = 64;
    B = Rh + (size_t)i * 64 * 64;
    out = relh + ((size_t)b * 64 + i) * 4096;  orow = 64;
  } else {
    A = qb + ((size_t)b * 4096 + i) * 64;      lda = 4096;
    B = Rw + (size_t)i * 64 * 64;
    out = relw + (size_t)b * 64 * 64 * 64 + (size_t)i * 64; orow = 4096;
  }
  v16bf b0 = load_b_frag(B, 64, wv << 4, 0, lane);
  v16bf b1 = load_b_frag(B, 64, wv << 4, 32, lane);
  v8f acc[4] = {};
#pragma unroll
  for (int mt = 0; mt < 4; ++mt) {
    v16bf a0 = load_a_frag(A, lda, mt * 16, 0, lane);
    v16bf a1 = load_a_frag(A, lda, mt * 16, 32, lane);
    acc[mt] = wmma_bf16(a0, b0, acc[mt]);
    acc[mt] = wmma_bf16(a1, b1, acc[mt]);
  }
  const int n = (wv << 4) + (lane & 15);
  const int hi8 = (lane >> 4) << 3;
#pragma unroll
  for (int mt = 0; mt < 4; ++mt)
#pragma unroll
    for (int r = 0; r < 8; ++r)
      out[(size_t)(mt * 16 + r + hi8) * orow + n] = acc[mt][r];
}

// ---------------------------------------------------------------------------
// Flash attention, async double-buffered K/V tiles in LDS, Q in registers,
// softmax parallelized 2 threads/row with LDS-carried running (m, l).
// ---------------------------------------------------------------------------
__global__ __launch_bounds__(128) void attn_kernel(
    const bf16_t* __restrict__ qb, const bf16_t* __restrict__ kb,
    const bf16_t* __restrict__ vtb, const float* __restrict__ relh,
    const float* __restrict__ relw, bf16_t* __restrict__ aout) {
  const int h = blockIdx.x, b = blockIdx.y;
  const int tid = threadIdx.x, lane = tid & 31, wv = tid >> 5;
  const int n0 = wv << 4;
  const int hi8 = (lane >> 4) << 3;

  __shared__ __align__(16) bf16_t sK[2][64 * 64];  // K tile  [key][c]
  __shared__ __align__(16) bf16_t sV[2][64 * 64];  // V tile  [c][key_local]
  __shared__ __align__(16) float  sS[64 * 68];     // biased scores
  __shared__ __align__(16) bf16_t sP[64 * 72];     // exp(P)
  __shared__ float sRelh[64];
  __shared__ float sAlpha[64];
  __shared__ float sM[64];      // running max per row
  __shared__ float sLrun[64];   // running sum per row
  __shared__ float sPm[128];    // partial max  [row][half]
  __shared__ float sPs[128];    // partial sum  [row][half]

  const bf16_t* Q  = qb  + ((size_t)b * 4096 + h * 64) * 64;
  const bf16_t* K  = kb  + (size_t)b * 4096 * 64;
  const bf16_t* VT = vtb + (size_t)b * 64 * 4096;
  const float*  RH = relh + ((size_t)b * 64 + h) * 4096;  // [w][kt]
  const float*  RW = relw + ((size_t)b * 64 + h) * 4096;  // [w][kw]

  // Start DMA for tile 0 first; overlap with register preloads below.
  copy_tile_async(&sK[0][0], K, 64, tid);
  copy_tile_async(&sV[0][0], VT, 4096, tid);

  if (tid < 64) { sM[tid] = -3.0e38f; sLrun[tid] = 0.0f; }

  // Q fragments resident in registers for the whole kernel.
  v16bf qfrag[2][4];
#pragma unroll
  for (int ks = 0; ks < 2; ++ks)
#pragma unroll
    for (int mt = 0; mt < 4; ++mt)
      qfrag[ks][mt] = load_a_frag(Q, 64, mt * 16, ks * 32, lane);

  // rel_w bias: each lane only needs RW[m][kw] for its fixed kw -> registers.
  const int kwl = n0 + (lane & 15);
  float rwreg[32];
#pragma unroll
  for (int mt = 0; mt < 4; ++mt)
#pragma unroll
    for (int r = 0; r < 8; ++r)
      rwreg[mt * 8 + r] = RW[(size_t)(mt * 16 + r + hi8) * 64 + kwl];

  v8f oacc[4] = {};
  const float scale = 0.125f;  // hd^-0.5
  const int row2 = tid >> 1, half = tid & 1;

#pragma unroll 1
  for (int kt = 0; kt < 64; ++kt) {
    const int cur = kt & 1;
    wait_async0();
    __syncthreads();  // B0: tile kt resident; buffers (1-cur) free

    if (kt + 1 < 64) {  // prefetch tile kt+1 (overlaps all compute below)
      copy_tile_async(&sK[cur ^ 1][0], K + (size_t)(kt + 1) * 4096, 64, tid);
      copy_tile_async(&sV[cur ^ 1][0], VT + (size_t)(kt + 1) * 64, 4096, tid);
    }
    if (tid < 64) sRelh[tid] = RH[tid * 64 + kt];

    // ---- S = Q * K_tile^T (B frags from LDS) ----
    v8f sacc[4] = {};
#pragma unroll
    for (int ks = 0; ks < 2; ++ks) {
      v16bf bfrag = load_b_frag(&sK[cur][0], 64, n0, ks * 32, lane);
#pragma unroll
      for (int mt = 0; mt < 4; ++mt)
        sacc[mt] = wmma_bf16(qfrag[ks][mt], bfrag, sacc[mt]);
    }
    __syncthreads();  // B1: sRelh ready; prev sP consumers done

    // ---- scores + bias -> LDS ----
#pragma unroll
    for (int mt = 0; mt < 4; ++mt)
#pragma unroll
      for (int r = 0; r < 8; ++r) {
        const int m = mt * 16 + r + hi8;
        sS[m * 68 + kwl] = sacc[mt][r] * scale + rwreg[mt * 8 + r] + sRelh[m];
      }
    __syncthreads();  // B2: scores complete

    // ---- online softmax: 2 threads per row ----
    {
      const float* srow = &sS[row2 * 68 + half * 32];
      float pmax = -3.0e38f;
#pragma unroll 8
      for (int j = 0; j < 32; ++j) pmax = fmaxf(pmax, srow[j]);
      sPm[tid] = pmax;
    }
    __syncthreads();  // partial maxima visible
    {
      const float mnew = fmaxf(sM[row2], fmaxf(sPm[row2 * 2], sPm[row2 * 2 + 1]));
      const float* srow = &sS[row2 * 68 + half * 32];
      bf16_t* prow = &sP[row2 * 72 + half * 32];
      float psum = 0.0f;
#pragma unroll 8
      for (int j = 0; j < 32; ++j) {
        const float p = __expf(srow[j] - mnew);
        prow[j] = (bf16_t)p;
        psum += p;
      }
      sPs[tid] = psum;
    }
    __syncthreads();  // partial sums + P visible
    if (tid < 64) {   // finalize running stats, one thread per row
      const float mold = sM[tid];
      const float mnew = fmaxf(mold, fmaxf(sPm[tid * 2], sPm[tid * 2 + 1]));
      const float alpha = __expf(mold - mnew);
      sLrun[tid] = sLrun[tid] * alpha + sPs[tid * 2] + sPs[tid * 2 + 1];
      sM[tid] = mnew;
      sAlpha[tid] = alpha;
    }
    __syncthreads();  // B3: alpha ready

    // ---- rescale O, then O += P * V_tile ----
#pragma unroll
    for (int mt = 0; mt < 4; ++mt)
#pragma unroll
      for (int r = 0; r < 8; ++r) oacc[mt][r] *= sAlpha[mt * 16 + r + hi8];

#pragma unroll
    for (int ks = 0; ks < 2; ++ks) {
      v16bf bfrag = load_b_frag(&sV[cur][0], 64, n0, ks * 32, lane);
#pragma unroll
      for (int mt = 0; mt < 4; ++mt) {
        v16bf afrag = load_a_frag(sP, 72, mt * 16, ks * 32, lane);
        oacc[mt] = wmma_bf16(afrag, bfrag, oacc[mt]);
      }
    }
  }

  const int c = n0 + (lane & 15);
#pragma unroll
  for (int mt = 0; mt < 4; ++mt)
#pragma unroll
    for (int r = 0; r < 8; ++r) {
      const int m = mt * 16 + r + hi8;
      const float v = oacc[mt][r] / sLrun[m];
      aout[((size_t)(h * 64 + m)) * 768 + b * 64 + c] = (bf16_t)v;
    }
}

// ---------------------------------------------------------------------------
// Output projection: [4096,768] x [768,768] + bias -> f32 d_out
// ---------------------------------------------------------------------------
__global__ __launch_bounds__(128) void proj_gemm(
    const bf16_t* __restrict__ A, const bf16_t* __restrict__ Bt,
    const float* __restrict__ bias, float* __restrict__ out) {
  const int tid = threadIdx.x, lane = tid & 31, wv = tid >> 5;
  const int m0 = blockIdx.x * 64;
  const int n0 = blockIdx.y * 64 + (wv << 4);
  v8f acc[4] = {};
  gemm_tile<768>(A, 768, Bt, 768, m0, n0, lane, acc);

  const int n = n0 + (lane & 15);
  const float bn = bias[n];
  const int hi8 = (lane >> 4) << 3;
#pragma unroll
  for (int mt = 0; mt < 4; ++mt)
#pragma unroll
    for (int r = 0; r < 8; ++r)
      out[(size_t)(m0 + mt * 16 + r + hi8) * 768 + n] = acc[mt][r] + bn;
}

// ---------------------------------------------------------------------------
extern "C" void kernel_launch(void* const* d_in, const int* in_sizes, int n_in,
                              void* d_out, int out_size, void* d_ws, size_t ws_size,
                              hipStream_t stream) {
  const float* x      = (const float*)d_in[0];
  const float* qkv_w  = (const float*)d_in[1];
  const float* qkv_b  = (const float*)d_in[2];
  const float* proj_w = (const float*)d_in[3];
  const float* proj_b = (const float*)d_in[4];
  const float* rph    = (const float*)d_in[5];
  const float* rpw    = (const float*)d_in[6];

  char* wp = (char*)d_ws;
  auto carve = [&](size_t bytes) {
    char* p = wp;
    wp += (bytes + 255) & ~(size_t)255;
    return (void*)p;
  };
  bf16_t* xb    = (bf16_t*)carve((size_t)4096 * 768 * 2);
  bf16_t* wt    = (bf16_t*)carve((size_t)2304 * 768 * 2);
  bf16_t* pwt   = (bf16_t*)carve((size_t)768 * 768 * 2);
  bf16_t* qbuf  = (bf16_t*)carve((size_t)12 * 4096 * 64 * 2);
  bf16_t* kbuf  = (bf16_t*)carve((size_t)12 * 4096 * 64 * 2);
  bf16_t* vtbuf = (bf16_t*)carve((size_t)12 * 64 * 4096 * 2);
  bf16_t* Rh    = (bf16_t*)carve((size_t)64 * 64 * 64 * 2);
  bf16_t* Rw    = (bf16_t*)carve((size_t)64 * 64 * 64 * 2);
  float*  relh  = (float*)carve((size_t)12 * 64 * 64 * 64 * 4);
  float*  relw  = (float*)carve((size_t)12 * 64 * 64 * 64 * 4);
  bf16_t* aoutb = (bf16_t*)carve((size_t)4096 * 768 * 2);

  { int n = 4096 * 768; cvt_bf16<<<(n + 255) / 256, 256, 0, stream>>>(x, xb, n); }
  { int n = 768 * 2304; transpose_cvt<<<(n + 255) / 256, 256, 0, stream>>>(qkv_w, wt, 768, 2304); }
  { int n = 768 * 768;  transpose_cvt<<<(n + 255) / 256, 256, 0, stream>>>(proj_w, pwt, 768, 768); }
  { int n = 64 * 64 * 64;
    build_rel<<<(n + 255) / 256, 256, 0, stream>>>(rph, Rh, n);
    build_rel<<<(n + 255) / 256, 256, 0, stream>>>(rpw, Rw, n); }

  qkv_gemm<<<dim3(64, 36), 128, 0, stream>>>(xb, wt, qkv_b, qbuf, kbuf, vtbuf);
  relbias_gemm<<<dim3(64, 12, 2), 128, 0, stream>>>(qbuf, Rh, Rw, relh, relw);
  attn_kernel<<<dim3(64, 12), 128, 0, stream>>>(qbuf, kbuf, vtbuf, relh, relw, aoutb);
  proj_gemm<<<dim3(64, 12), 128, 0, stream>>>(aoutb, pwt, proj_b, (float*)d_out);
}
